// EncoderBlock_44470091382845
// MI455X (gfx1250) — matrix operand order repaired and checked by reading, not compiled, not run on hardware
//
#include <hip/hip_runtime.h>

typedef __attribute__((ext_vector_type(16))) _Float16 v16h;
typedef __attribute__((ext_vector_type(8)))  _Float16 v8h;
typedef __attribute__((ext_vector_type(8)))  float    v8f;
typedef __attribute__((ext_vector_type(4)))  int      v4i;

#define EMB   768
#define HEADS 8
#define HD    96
#define DFF   3072
#define SEQ   2048
#define BATCH 4
#define MROWS (BATCH * SEQ)

#if __has_builtin(__builtin_amdgcn_global_load_async_to_lds_b128) && \
    __has_builtin(__builtin_amdgcn_s_wait_asynccnt)
#define USE_ASYNC_LDS 1
#endif

#ifdef USE_ASYNC_LDS
typedef __attribute__((address_space(1))) v4i* gv4i_p;  // global v4i*
typedef __attribute__((address_space(3))) v4i* lv4i_p;  // LDS v4i*
#define GLOAD_ASYNC_B128(dst, src)                                   \
  __builtin_amdgcn_global_load_async_to_lds_b128(                    \
      (gv4i_p)(src), (lv4i_p)(dst), 0, 0)
#endif

__device__ __forceinline__ v16h cat16(v8h lo, v8h hi) {
  return __builtin_shufflevector(lo, hi, 0,1,2,3,4,5,6,7,8,9,10,11,12,13,14,15);
}

__device__ __forceinline__ v8f wmma16(v16h a, v16h b, v8f c) {
  // D = A(16x32 f16) x B(32x16 f16) + C(16x16 f32)
  return __builtin_amdgcn_wmma_f32_16x16x32_f16(false, a, false, b, (short)0, c,
                                                false, false);
}

// ---------------------------------------------------------------------------
// Weight cast+transpose: Wt[n*K + k] = (f16) W[k*N + n]
// ---------------------------------------------------------------------------
__global__ void cast_transpose_kernel(const float* __restrict__ W,
                                      _Float16* __restrict__ Wt, int K, int N) {
  int idx = blockIdx.x * 256 + threadIdx.x;
  if (idx >= K * N) return;
  int n = idx / K;
  int k = idx - n * K;
  Wt[idx] = (_Float16)W[(size_t)k * N + n];
}

// ---------------------------------------------------------------------------
// LayerNorm over last dim (768) + cast to f16. One block per row.
// ---------------------------------------------------------------------------
__global__ __launch_bounds__(256) void ln_cast_kernel(
    const float* __restrict__ x, const float* __restrict__ g,
    const float* __restrict__ bta, _Float16* __restrict__ out) {
  int row = blockIdx.x;
  int tid = threadIdx.x;
  const float* xr = x + (size_t)row * EMB;
  float v0 = xr[tid], v1 = xr[tid + 256], v2 = xr[tid + 512];
  float s1 = v0 + v1 + v2;
  float s2 = v0 * v0 + v1 * v1 + v2 * v2;
  for (int m = 16; m >= 1; m >>= 1) {
    s1 += __shfl_xor(s1, m, 32);
    s2 += __shfl_xor(s2, m, 32);
  }
  __shared__ float a1[8], a2[8];
  int w = tid >> 5, lane = tid & 31;
  if (lane == 0) { a1[w] = s1; a2[w] = s2; }
  __syncthreads();
  s1 = 0.f; s2 = 0.f;
  for (int i = 0; i < 8; ++i) { s1 += a1[i]; s2 += a2[i]; }
  float mu  = s1 * (1.0f / EMB);
  float var = s2 * (1.0f / EMB) - mu * mu;
  float rs  = rsqrtf(var + 1e-5f);
  _Float16* orow = out + (size_t)row * EMB;
  orow[tid]       = (_Float16)((v0 - mu) * rs * g[tid]       + bta[tid]);
  orow[tid + 256] = (_Float16)((v1 - mu) * rs * g[tid + 256] + bta[tid + 256]);
  orow[tid + 512] = (_Float16)((v2 - mu) * rs * g[tid + 512] + bta[tid + 512]);
}

// ---------------------------------------------------------------------------
// WMMA GEMM: C[M,N] = A[M,K](f16) * Bt[N,K]^T(f16) + bias (+gelu) (+resid)
// Block tile 256x64, BK=32, double-buffered LDS fed by async global->LDS.
// 8 waves, each computes 32x64 (2 slabs x 4 accums): each B-fragment feeds
// two WMMAs. Stage i+1 streams in while stage i computes; deferred
// s_wait_asynccnt exploits in-order async completion.
// flags: bit0 = exact GELU, bit1 = head-scatter f16 output [b,h,n,d]
// ---------------------------------------------------------------------------
__global__ __launch_bounds__(256) void gemm_wmma_kernel(
    const _Float16* __restrict__ A, const _Float16* __restrict__ Bt,
    const float* __restrict__ bias, const float* __restrict__ resid,
    float* __restrict__ outF, _Float16* __restrict__ outH,
    int M, int N, int K, int flags) {
  __shared__ __align__(16) _Float16 As[2][256][40];  // padded rows (80B) vs bank conflicts
  __shared__ __align__(16) _Float16 Bs[2][64][40];

  int tid  = threadIdx.x;
  int w    = tid >> 5;
  int lane = tid & 31;
  int half = lane >> 4;
  int lrow = lane & 15;
  int rowBase = blockIdx.y * 256;
  int colBase = blockIdx.x * 64;

  v8f acc[2][4];
  for (int s = 0; s < 2; ++s)
    for (int i = 0; i < 4; ++i)
      acc[s][i] = v8f{0.f,0.f,0.f,0.f,0.f,0.f,0.f,0.f};

  // loader assignments: A = one 32-f16 row slice per thread; B = half row per
  // thread for tid<128 (waves 0-3 issue 6 async copies/stage, waves 4-7: 4)
  const _Float16* aRow = A + (size_t)(rowBase + tid) * K;
  int br  = tid >> 1;
  int bh2 = tid & 1;
  const _Float16* bRow = Bt + (size_t)(colBase + br) * K + bh2 * 16;

  auto issue_stage = [&](int kk, int buf) {
#ifdef USE_ASYNC_LDS
#pragma unroll
    for (int u = 0; u < 4; ++u)
      GLOAD_ASYNC_B128(&As[buf][tid][u * 8], aRow + kk + u * 8);
    if (tid < 128) {
      GLOAD_ASYNC_B128(&Bs[buf][br][bh2 * 16],     bRow + kk);
      GLOAD_ASYNC_B128(&Bs[buf][br][bh2 * 16 + 8], bRow + kk + 8);
    }
#else
#pragma unroll
    for (int u = 0; u < 4; ++u)
      *(v8h*)&As[buf][tid][u * 8] = *(const v8h*)(aRow + kk + u * 8);
    if (tid < 128) {
      *(v8h*)&Bs[buf][br][bh2 * 16]     = *(const v8h*)(bRow + kk);
      *(v8h*)&Bs[buf][br][bh2 * 16 + 8] = *(const v8h*)(bRow + kk + 8);
    }
#endif
  };

  int koff = half * 8;  // A-fragment K split per ISA 16-bit A layout
  int r0   = w * 32;

  issue_stage(0, 0);  // prologue

  int cur = 0;
  for (int k0 = 0; k0 < K; k0 += 32, cur ^= 1) {
    if (k0 + 32 < K) {
      issue_stage(k0 + 32, cur ^ 1);  // overlap next stage with this compute
#ifdef USE_ASYNC_LDS
      // async copies complete in order: asynccnt <= (newest stage count)
      // guarantees the current stage has fully landed in LDS.
      if (tid < 128) __builtin_amdgcn_s_wait_asynccnt(6);
      else           __builtin_amdgcn_s_wait_asynccnt(4);
#endif
    } else {
#ifdef USE_ASYNC_LDS
      __builtin_amdgcn_s_wait_asynccnt(0);
#endif
    }
    __syncthreads();  // current stage visible to all waves

    v16h af0 = cat16(*(const v8h*)&As[cur][r0 + lrow][koff],
                     *(const v8h*)&As[cur][r0 + lrow][koff + 16]);
    v16h af1 = cat16(*(const v8h*)&As[cur][r0 + 16 + lrow][koff],
                     *(const v8h*)&As[cur][r0 + 16 + lrow][koff + 16]);
#pragma unroll
    for (int i = 0; i < 4; ++i) {
      v16h bf = cat16(*(const v8h*)&Bs[cur][i * 16 + lrow][half * 8],
                      *(const v8h*)&Bs[cur][i * 16 + lrow][half * 8 + 16]);
      acc[0][i] = wmma16(af0, bf, acc[0][i]);
      acc[1][i] = wmma16(af1, bf, acc[1][i]);
    }
    __syncthreads();  // all reads of 'cur' done before it is refilled
  }

  // Epilogue: C layout -> row = 8*half + j, col = lrow (per 16x16 tile)
#pragma unroll
  for (int s = 0; s < 2; ++s) {
#pragma unroll
    for (int i = 0; i < 4; ++i) {
      int col = colBase + i * 16 + lrow;
      float bv = bias ? bias[col] : 0.0f;
#pragma unroll
      for (int j = 0; j < 8; ++j) {
        int row = rowBase + w * 32 + s * 16 + half * 8 + j;
        float v = acc[s][i][j] + bv;
        if (flags & 1) v = 0.5f * v * (1.0f + erff(v * 0.70710678118f));  // exact gelu
        if (resid) v += resid[(size_t)row * N + col];
        if (outF) outF[(size_t)row * N + col] = v;
        if (outH) {
          size_t idx;
          if (flags & 2) {  // scatter to [b, h, n, d]
            int b = row / SEQ, n = row - b * SEQ;
            int h = col / HD,  d = col - h * HD;
            idx = ((size_t)(b * HEADS + h) * SEQ + n) * HD + d;
          } else {
            idx = (size_t)row * N + col;
          }
          outH[idx] = (_Float16)v;
        }
      }
    }
  }
}

// ---------------------------------------------------------------------------
// Flash attention per (b,h). 128 query rows per block (16 per wave),
// kv chunks of 64. softmax FIRST, then /sqrt(EMB) folded into final scale.
// ---------------------------------------------------------------------------
__global__ __launch_bounds__(256) void attn_wmma_kernel(
    const _Float16* __restrict__ qg, const _Float16* __restrict__ kg,
    const _Float16* __restrict__ vg, _Float16* __restrict__ og) {
  __shared__ __align__(16) _Float16 Vt[96][72];      // V^T (d-major), padded
  __shared__ __align__(16) _Float16 Pl[8][16][72];   // per-wave P staging

  int tid  = threadIdx.x;
  int w    = tid >> 5;
  int lane = tid & 31;
  int half = lane >> 4;
  int lrow = lane & 15;

  int bh = blockIdx.y;
  int b  = bh / HEADS, h = bh - b * HEADS;
  const _Float16* q = qg + (size_t)bh * SEQ * HD;
  const _Float16* k = kg + (size_t)bh * SEQ * HD;
  const _Float16* v = vg + (size_t)bh * SEQ * HD;

  int q0 = blockIdx.x * 128;

  // Q a-fragments (16x96, 3 k-steps of 32) held in registers
  v16h qa[3];
  {
    int qr = q0 + w * 16 + lrow;
#pragma unroll
    for (int s = 0; s < 3; ++s) {
      const _Float16* src = q + (size_t)qr * HD + s * 32 + half * 8;
      qa[s] = cat16(*(const v8h*)src, *(const v8h*)(src + 16));
    }
  }

  v8f o[6];
  for (int i = 0; i < 6; ++i) o[i] = v8f{0.f,0.f,0.f,0.f,0.f,0.f,0.f,0.f};
  float mi[8], li[8];
#pragma unroll
  for (int j = 0; j < 8; ++j) { mi[j] = -3.0e38f; li[j] = 0.f; }

  for (int kv0 = 0; kv0 < SEQ; kv0 += 64) {
    __syncthreads();  // protect Vt reuse across iterations
    {   // cooperative transpose-load of V chunk (64 x 96) into Vt
      int r = tid >> 2;            // kv row 0..63
      int d0 = (tid & 3) * 24;     // 24 d-elements each
#pragma unroll
      for (int u = 0; u < 3; ++u) {
        v8h vv = *(const v8h*)(v + (size_t)(kv0 + r) * HD + d0 + u * 8);
#pragma unroll
        for (int e = 0; e < 8; ++e) Vt[d0 + u * 8 + e][r] = vv[e];
      }
    }

    // S = Q * K^T  (16 x 64 per wave): K rows are contiguous = B^T layout
    v8f s[4];
    for (int i = 0; i < 4; ++i) s[i] = v8f{0.f,0.f,0.f,0.f,0.f,0.f,0.f,0.f};
#pragma unroll
    for (int t = 0; t < 4; ++t) {
#pragma unroll
      for (int s2 = 0; s2 < 3; ++s2) {
        const _Float16* src =
            k + (size_t)(kv0 + t * 16 + lrow) * HD + s2 * 32 + half * 8;
        v16h bf = cat16(*(const v8h*)src, *(const v8h*)(src + 16));
        s[t] = wmma16(qa[s2], bf, s[t]);
      }
    }

    // online softmax (row stats: 16-lane shuffle reductions within half-wave)
    float mnew[8], scale[8];
#pragma unroll
    for (int j = 0; j < 8; ++j) {
      float m = fmaxf(fmaxf(s[0][j], s[1][j]), fmaxf(s[2][j], s[3][j]));
      for (int msk = 8; msk >= 1; msk >>= 1) m = fmaxf(m, __shfl_xor(m, msk, 32));
      mnew[j]  = fmaxf(mi[j], m);
      scale[j] = __expf(mi[j] - mnew[j]);
    }
#pragma unroll
    for (int j = 0; j < 8; ++j) {
      float acc = 0.f;
#pragma unroll
      for (int t = 0; t < 4; ++t) {
        float p = __expf(s[t][j] - mnew[j]);
        s[t][j] = p;
        acc += p;
      }
      for (int msk = 8; msk >= 1; msk >>= 1) acc += __shfl_xor(acc, msk, 32);
      li[j] = li[j] * scale[j] + acc;
      mi[j] = mnew[j];
    }
#pragma unroll
    for (int dt = 0; dt < 6; ++dt)
#pragma unroll
      for (int j = 0; j < 8; ++j) o[dt][j] *= scale[j];

    // stage P (f16) per wave for A-fragment reload
#pragma unroll
    for (int t = 0; t < 4; ++t)
#pragma unroll
      for (int j = 0; j < 8; ++j)
        Pl[w][half * 8 + j][t * 16 + lrow] = (_Float16)s[t][j];
    __syncthreads();  // Vt fully written before use

    // O += P * V : K = 64 (2 wmma k-steps), 6 d-tiles
#pragma unroll
    for (int s2 = 0; s2 < 2; ++s2) {
      const _Float16* pp = &Pl[w][lrow][s2 * 32 + half * 8];
      v16h pa = cat16(*(const v8h*)pp, *(const v8h*)(pp + 16));
#pragma unroll
      for (int dt = 0; dt < 6; ++dt) {
        const _Float16* vb = &Vt[dt * 16 + lrow][s2 * 32 + half * 8];
        o[dt] = wmma16(pa, cat16(*(const v8h*)vb, *(const v8h*)(vb + 16)), o[dt]);
      }
    }
  }

  // normalize: softmax / sqrt(EMB) (reference order) then write [row, emb] f16
  float inv = 1.0f / sqrtf((float)EMB);
#pragma unroll
  for (int j = 0; j < 8; ++j) {
    float f = inv / li[j];
    int qrow = q0 + w * 16 + half * 8 + j;
    _Float16* orow = og + ((size_t)(b * SEQ + qrow)) * EMB + h * HD;
#pragma unroll
    for (int dt = 0; dt < 6; ++dt)
      orow[dt * 16 + lrow] = (_Float16)(o[dt][j] * f);
  }
}

// ---------------------------------------------------------------------------
extern "C" void kernel_launch(void* const* d_in, const int* in_sizes, int n_in,
                              void* d_out, int out_size, void* d_ws, size_t ws_size,
                              hipStream_t stream) {
  (void)in_sizes; (void)n_in; (void)out_size; (void)ws_size;
  const float* x    = (const float*)d_in[0];
  const float* g1   = (const float*)d_in[1];
  const float* be1  = (const float*)d_in[2];
  const float* wq   = (const float*)d_in[3];
  const float* bq   = (const float*)d_in[4];
  const float* wk   = (const float*)d_in[5];
  const float* bk   = (const float*)d_in[6];
  const float* wv   = (const float*)d_in[7];
  const float* bv   = (const float*)d_in[8];
  const float* wo   = (const float*)d_in[9];
  const float* bo   = (const float*)d_in[10];
  const float* g2   = (const float*)d_in[11];
  const float* be2  = (const float*)d_in[12];
  const float* w1   = (const float*)d_in[13];
  const float* bf1  = (const float*)d_in[14];
  const float* w2   = (const float*)d_in[15];
  const float* bf2  = (const float*)d_in[16];

  char* ws = (char*)d_ws;
  size_t off = 0;
  auto carve = [&](size_t bytes) -> char* {
    char* p = ws + off;
    off += (bytes + 255) & ~(size_t)255;
    return p;
  };
  _Float16* wqT = (_Float16*)carve((size_t)EMB * EMB * 2);
  _Float16* wkT = (_Float16*)carve((size_t)EMB * EMB * 2);
  _Float16* wvT = (_Float16*)carve((size_t)EMB * EMB * 2);
  _Float16* woT = (_Float16*)carve((size_t)EMB * EMB * 2);
  _Float16* w1T = (_Float16*)carve((size_t)DFF * EMB * 2);
  _Float16* w2T = (_Float16*)carve((size_t)EMB * DFF * 2);
  _Float16* yh  = (_Float16*)carve((size_t)MROWS * EMB * 2);  // ln1 out; reused for ln2 out
  _Float16* qh  = (_Float16*)carve((size_t)MROWS * EMB * 2);
  _Float16* kh  = (_Float16*)carve((size_t)MROWS * EMB * 2);
  _Float16* vh  = (_Float16*)carve((size_t)MROWS * EMB * 2);
  _Float16* oh  = (_Float16*)carve((size_t)MROWS * EMB * 2);
  float*    x1  = (float*)   carve((size_t)MROWS * EMB * 4);
  _Float16* hh  = qh;  // qh..oh contiguous = exactly MROWS*DFF*2 bytes; reuse for FFN hidden

  // 1) weight cast+transpose to f16 [N][K]
  cast_transpose_kernel<<<(EMB * EMB + 255) / 256, 256, 0, stream>>>(wq, wqT, EMB, EMB);
  cast_transpose_kernel<<<(EMB * EMB + 255) / 256, 256, 0, stream>>>(wk, wkT, EMB, EMB);
  cast_transpose_kernel<<<(EMB * EMB + 255) / 256, 256, 0, stream>>>(wv, wvT, EMB, EMB);
  cast_transpose_kernel<<<(EMB * EMB + 255) / 256, 256, 0, stream>>>(wo, woT, EMB, EMB);
  cast_transpose_kernel<<<(EMB * DFF + 255) / 256, 256, 0, stream>>>(w1, w1T, EMB, DFF);
  cast_transpose_kernel<<<(DFF * EMB + 255) / 256, 256, 0, stream>>>(w2, w2T, DFF, EMB);

  // 2) LN1
  ln_cast_kernel<<<MROWS, 256, 0, stream>>>(x, g1, be1, yh);

  // 3) QKV projections (head-scatter output)
  dim3 g768(EMB / 64, MROWS / 256);
  gemm_wmma_kernel<<<g768, 256, 0, stream>>>(yh, wqT, bq, nullptr, nullptr, qh,
                                             MROWS, EMB, EMB, 2);
  gemm_wmma_kernel<<<g768, 256, 0, stream>>>(yh, wkT, bk, nullptr, nullptr, kh,
                                             MROWS, EMB, EMB, 2);
  gemm_wmma_kernel<<<g768, 256, 0, stream>>>(yh, wvT, bv, nullptr, nullptr, vh,
                                             MROWS, EMB, EMB, 2);

  // 4) attention
  dim3 ga(SEQ / 128, BATCH * HEADS);
  attn_wmma_kernel<<<ga, 256, 0, stream>>>(qh, kh, vh, oh);

  // 5) output projection + residual -> x1 (f32)
  gemm_wmma_kernel<<<g768, 256, 0, stream>>>(oh, woT, bo, x, x1, nullptr,
                                             MROWS, EMB, EMB, 0);

  // 6) LN2 -> yh (reused)
  ln_cast_kernel<<<MROWS, 256, 0, stream>>>(x1, g2, be2, yh);

  // 7) FFN1 + exact gelu -> hh (f16)
  dim3 gffn(DFF / 64, MROWS / 256);
  gemm_wmma_kernel<<<gffn, 256, 0, stream>>>(yh, w1T, bf1, nullptr, nullptr, hh,
                                             MROWS, DFF, EMB, 1);

  // 8) FFN2 + bias + residual -> d_out (f32)
  gemm_wmma_kernel<<<g768, 256, 0, stream>>>(hh, w2T, bf2, x1, (float*)d_out,
                                             nullptr, MROWS, EMB, DFF, 0);
}
